// LSTM_35304631173873
// MI455X (gfx1250) — compile-verified, matrix operand order checked by reading, not run on hardware
//
#include <hip/hip_runtime.h>

// ---------------------------------------------------------------------------
// 2-layer LSTM (H1=80, H2=100), B=1024, T=512, + MLP head on last timestep.
//
// Strategy (MI455X / gfx1250, wave32, WMMA):
//  * Recurrence is independent across batch rows -> each wave owns a 16-row
//    batch tile persistently across all 512 timesteps (no global sync).
//  * Per step, gates g = [in , h] @ [W_in ; W_hh]^T computed with
//    v_wmma_f32_16x16x32_f16; weights live in LDS in B-fragment layout (f16).
//  * Layer-1 writes relu(h1) as f16 (84 MB, fits in 192 MB L2); layer-2
//    streams it back per step and fuses the input projection into the same
//    K-concatenated WMMA loop. Only h2[:, T-1] is kept (head fused in-kernel).
// ---------------------------------------------------------------------------

typedef __attribute__((ext_vector_type(16))) _Float16 v16h;
typedef __attribute__((ext_vector_type(8)))  float    v8f;

#define TLEN 512

__device__ __forceinline__ float fsig(float x) {
  return __builtin_amdgcn_rcpf(1.0f + __expf(-x));          // v_exp + v_rcp
}
__device__ __forceinline__ float ftanh_(float x) {
  return 2.0f * __builtin_amdgcn_rcpf(1.0f + __expf(-2.0f * x)) - 1.0f;
}
__device__ __forceinline__ float frelu(float x) { return x > 0.0f ? x : 0.0f; }

// B fragment: 32 lanes x 32 contiguous bytes (1 KB / fragment) in LDS.
__device__ __forceinline__ v16h ld_bfrag(const char* p) {
  union { v16h v; uint4 q[2]; } u;
  u.q[0] = *(const uint4*)(p);
  u.q[1] = *(const uint4*)(p + 16);
  return u.v;
}
// A fragment from row-major staging: v0-3 at k=8*half, v4-7 at k=16+8*half.
__device__ __forceinline__ v16h ld_afrag(const char* p) {
  union { v16h v; uint4 q[2]; } u;
  u.q[0] = *(const uint4*)(p);
  u.q[1] = *(const uint4*)(p + 32);
  return u.v;
}

extern __shared__ char smem[];

// ===========================================================================
// Layer-1 LSTM: input dim 1, H=80. K = 1 + 80 -> padded 96 (3 k-tiles).
// N = 4*80 = 320 (20 n-tiles, gate boundaries 16-aligned).
// 4 waves/block, each wave = 16 batch rows, grid = 16 -> B = 1024.
// ===========================================================================
__global__ __launch_bounds__(128, 1)
void lstm_l1_kernel(const float* __restrict__ x,
                    const float* __restrict__ Wih, const float* __restrict__ Whh,
                    const float* __restrict__ bih, const float* __restrict__ bhh,
                    _Float16* __restrict__ h1out) {
  constexpr int KT = 3, NT = 20, NTH = 5;
  constexpr int W_ELEMS = KT * NT * 512;          // f16 elements in fragments
  constexpr int BIAS_OFF = KT * NT * 1024;        // 61440
  constexpr int A_OFF = BIAS_OFF + 320 * 4;       // 62720
  constexpr int AW = 16 * 192;                    // A staging per wave (K=96 f16)
  constexpr int O_OFF = A_OFF + 4 * AW;           // 75008
  constexpr int OW = 16 * 160;                    // relu(h) staging (80 f16/row)

  const int tid = threadIdx.x, wave = tid >> 5, lane = tid & 31;
  const int half = lane >> 4, nl = lane & 15;
  const int b0 = blockIdx.x * 64 + wave * 16;

  // ---- stage [W_ih ; W_hh] into f16 B-fragment layout (once) ----
  for (int idx = tid; idx < W_ELEMS; idx += 128) {
    int frag = idx >> 9, r = idx & 511;
    int fl = r >> 4, j = r & 15;
    int v = j >> 1, pair = j & 1;
    int kl = (v < 4 ? 0 : 16) + 8 * (fl >> 4) + 2 * (v & 3) + pair;
    int k = (frag / NT) * 32 + kl;
    int n = (frag % NT) * 16 + (fl & 15);
    float wv = 0.0f;
    if (k == 0)        wv = Wih[n];               // input column
    else if (k <= 80)  wv = Whh[n * 80 + (k - 1)];
    ((_Float16*)smem)[(size_t)frag * 512 + fl * 16 + j] = (_Float16)wv;
  }
  for (int n = tid; n < 320; n += 128)
    *(float*)(smem + BIAS_OFF + n * 4) = bih[n] + bhh[n];
  {
    char* Aw0 = smem + A_OFF + wave * AW;         // zero h + pad cols (h0 = 0)
    for (int i = lane; i < AW / 4; i += 32) ((unsigned*)Aw0)[i] = 0u;
  }
  __syncthreads();

  char* Aw = smem + A_OFF + wave * AW;
  char* Ow = smem + O_OFF + wave * OW;

  int crow[5], coff[5];
  _Float16* gout[5];
#pragma unroll
  for (int i = 0; i < 5; ++i) {                   // coalesced 16B store chunks
    int c0 = lane + 32 * i;
    crow[i] = c0 / 10; coff[i] = c0 % 10;
    gout[i] = h1out + (size_t)(b0 + crow[i]) * TLEN * 80 + coff[i] * 8;
  }
  const float* xrow = x + (size_t)(b0 + nl) * TLEN;

  v8f cc[NTH] = {};                               // cell state tiles (f32)

#pragma unroll 1
  for (int t = 0; t < TLEN; ++t) {
    if (lane < 16) *(_Float16*)(Aw + lane * 192) = (_Float16)xrow[t];  // k=0
    asm volatile("s_wait_dscnt 0x0" ::: "memory");

    v16h a[KT];
#pragma unroll
    for (int kt = 0; kt < KT; ++kt)
      a[kt] = ld_afrag(Aw + nl * 192 + kt * 64 + 16 * half);

#pragma unroll
    for (int ht = 0; ht < NTH; ++ht) {
      v8f acc[4];
#pragma unroll
      for (int g = 0; g < 4; ++g) {
        float bv = *(const float*)(smem + BIAS_OFF + (g * 80 + ht * 16 + nl) * 4);
        acc[g] = (v8f){bv, bv, bv, bv, bv, bv, bv, bv};
#pragma unroll
        for (int kt = 0; kt < KT; ++kt) {
          const char* bp = smem + (size_t)(kt * NT + g * NTH + ht) * 1024 + lane * 32;
          acc[g] = __builtin_amdgcn_wmma_f32_16x16x32_f16(
              false, a[kt], false, ld_bfrag(bp), (short)0, acc[g], false, false);
        }
      }
#pragma unroll
      for (int e = 0; e < 8; ++e) {
        float iv = fsig(acc[0][e]);
        float fv = fsig(acc[1][e]);
        float gv = ftanh_(acc[2][e]);
        float ov = fsig(acc[3][e]);
        float cn = fv * cc[ht][e] + iv * gv;
        cc[ht][e] = cn;
        float hv = ov * ftanh_(cn);
        int m = e + 8 * half;
        *(_Float16*)(Aw + m * 192 + (1 + ht * 16 + nl) * 2) = (_Float16)hv;       // raw h (recurrence)
        *(_Float16*)(Ow + m * 160 + (ht * 16 + nl) * 2) = (_Float16)frelu(hv);    // relu(h) -> layer 2
      }
    }
    asm volatile("s_wait_dscnt 0x0" ::: "memory");
#pragma unroll
    for (int i = 0; i < 5; ++i) {                 // coalesced b128 global stores
      uint4 d = *(const uint4*)(Ow + crow[i] * 160 + coff[i] * 16);
      *(uint4*)(gout[i] + (size_t)t * 80) = d;
    }
  }
}

// ===========================================================================
// Layer-2 LSTM + head: K = 80 (h1_t) + 100 (h2) -> padded 192 (6 k-tiles).
// N = 4 gates x H padded to 112 = 448 (28 n-tiles). Padded hidden lanes get
// zero weights/bias -> c=0 -> h=0 automatically.
// ===========================================================================
__global__ __launch_bounds__(128, 1)
void lstm_l2_head_kernel(const _Float16* __restrict__ h1in,
                         const float* __restrict__ Wih, const float* __restrict__ Whh,
                         const float* __restrict__ bih, const float* __restrict__ bhh,
                         const float* __restrict__ Wl1, const float* __restrict__ bl1,
                         const float* __restrict__ Wl2, const float* __restrict__ bl2,
                         float* __restrict__ out) {
  constexpr int KT = 6, NT = 28, NTH = 7;
  constexpr int W_ELEMS = KT * NT * 512;          // 86016 f16
  constexpr int BIAS_OFF = KT * NT * 1024;        // 172032
  constexpr int A_OFF = BIAS_OFF + 448 * 4;       // 173824
  constexpr int AW = 16 * 384;                    // 16 rows x 192 f16

  const int tid = threadIdx.x, wave = tid >> 5, lane = tid & 31;
  const int half = lane >> 4, nl = lane & 15;
  const int b0 = blockIdx.x * 64 + wave * 16;

  for (int idx = tid; idx < W_ELEMS; idx += 128) {
    int frag = idx >> 9, r = idx & 511;
    int fl = r >> 4, j = r & 15;
    int v = j >> 1, pair = j & 1;
    int kl = (v < 4 ? 0 : 16) + 8 * (fl >> 4) + 2 * (v & 3) + pair;
    int k = (frag / NT) * 32 + kl;
    int n = (frag % NT) * 16 + (fl & 15);
    int gate = n / 112, jj = n % 112;
    float wv = 0.0f;
    if (jj < 100) {
      int row = gate * 100 + jj;
      if (k < 80)       wv = Wih[row * 80 + k];
      else if (k < 180) wv = Whh[row * 100 + (k - 80)];
    }
    ((_Float16*)smem)[(size_t)frag * 512 + fl * 16 + j] = (_Float16)wv;
  }
  for (int n = tid; n < 448; n += 128) {
    int gate = n / 112, jj = n % 112;
    *(float*)(smem + BIAS_OFF + n * 4) =
        (jj < 100) ? (bih[gate * 100 + jj] + bhh[gate * 100 + jj]) : 0.0f;
  }
  {
    char* Aw0 = smem + A_OFF + wave * AW;
    for (int i = lane; i < AW / 4; i += 32) ((unsigned*)Aw0)[i] = 0u;
  }
  __syncthreads();

  char* Aw = smem + A_OFF + wave * AW;
  int crow[5], coff[5];
  const _Float16* gin[5];
#pragma unroll
  for (int i = 0; i < 5; ++i) {
    int c0 = lane + 32 * i;
    crow[i] = c0 / 10; coff[i] = c0 % 10;
    gin[i] = h1in + (size_t)(b0 + crow[i]) * TLEN * 80 + coff[i] * 8;
  }

  v8f cc[NTH] = {};

#pragma unroll 1
  for (int t = 0; t < TLEN; ++t) {
#pragma unroll
    for (int i = 0; i < 5; ++i) {                 // stream h1_t from L2 cache
      uint4 d = *(const uint4*)(gin[i] + (size_t)t * 80);
      *(uint4*)(Aw + crow[i] * 384 + coff[i] * 16) = d;
      if (t + 1 < TLEN) __builtin_prefetch(gin[i] + (size_t)(t + 1) * 80, 0, 1);
    }
    asm volatile("s_wait_dscnt 0x0" ::: "memory");

    v16h a[KT];
#pragma unroll
    for (int kt = 0; kt < KT; ++kt)
      a[kt] = ld_afrag(Aw + nl * 384 + kt * 64 + 16 * half);

#pragma unroll
    for (int ht = 0; ht < NTH; ++ht) {
      v8f acc[4];
#pragma unroll
      for (int g = 0; g < 4; ++g) {
        float bv = *(const float*)(smem + BIAS_OFF + (g * 112 + ht * 16 + nl) * 4);
        acc[g] = (v8f){bv, bv, bv, bv, bv, bv, bv, bv};
#pragma unroll
        for (int kt = 0; kt < KT; ++kt) {
          const char* bp = smem + (size_t)(kt * NT + g * NTH + ht) * 1024 + lane * 32;
          acc[g] = __builtin_amdgcn_wmma_f32_16x16x32_f16(
              false, a[kt], false, ld_bfrag(bp), (short)0, acc[g], false, false);
        }
      }
#pragma unroll
      for (int e = 0; e < 8; ++e) {
        float iv = fsig(acc[0][e]);
        float fv = fsig(acc[1][e]);
        float gv = ftanh_(acc[2][e]);
        float ov = fsig(acc[3][e]);
        float cn = fv * cc[ht][e] + iv * gv;
        cc[ht][e] = cn;
        float hv = ov * ftanh_(cn);
        int m = e + 8 * half;
        *(_Float16*)(Aw + m * 384 + (80 + ht * 16 + nl) * 2) = (_Float16)hv;
      }
    }
  }

  asm volatile("s_wait_dscnt 0x0" ::: "memory");
  // ---- fused MLP head on relu(h2[:, T-1]) : lanes 0..15, one row each ----
  if (lane < 16) {
    const int m = lane;
    float z[10];
#pragma unroll
    for (int q = 0; q < 10; ++q) z[q] = bl1[q];
    for (int j = 0; j < 100; ++j) {
      float hv = frelu((float)*(const _Float16*)(Aw + m * 384 + (80 + j) * 2));
#pragma unroll
      for (int q = 0; q < 10; ++q) z[q] += hv * Wl1[q * 100 + j];
    }
    float o = bl2[0];
#pragma unroll
    for (int q = 0; q < 10; ++q) o += frelu(z[q]) * Wl2[q];
    out[b0 + m] = o;
  }
}

// ===========================================================================
extern "C" void kernel_launch(void* const* d_in, const int* in_sizes, int n_in,
                              void* d_out, int out_size, void* d_ws, size_t ws_size,
                              hipStream_t stream) {
  const float* x    = (const float*)d_in[0];
  const float* Wih1 = (const float*)d_in[1];
  const float* Whh1 = (const float*)d_in[2];
  const float* bih1 = (const float*)d_in[3];
  const float* bhh1 = (const float*)d_in[4];
  const float* Wih2 = (const float*)d_in[5];
  const float* Whh2 = (const float*)d_in[6];
  const float* bih2 = (const float*)d_in[7];
  const float* bhh2 = (const float*)d_in[8];
  const float* Wl1  = (const float*)d_in[9];
  const float* bl1  = (const float*)d_in[10];
  const float* Wl2  = (const float*)d_in[11];
  const float* bl2  = (const float*)d_in[12];

  _Float16* h1buf = (_Float16*)d_ws;              // 1024*512*80 f16 = 83.9 MB

  constexpr int SMEM1 = 3 * 20 * 1024 + 320 * 4 + 4 * (16 * 192) + 4 * (16 * 160); // 85248
  constexpr int SMEM2 = 6 * 28 * 1024 + 448 * 4 + 4 * (16 * 384);                  // 198400

  (void)hipFuncSetAttribute((const void*)lstm_l1_kernel,
                            hipFuncAttributeMaxDynamicSharedMemorySize, SMEM1);
  (void)hipFuncSetAttribute((const void*)lstm_l2_head_kernel,
                            hipFuncAttributeMaxDynamicSharedMemorySize, SMEM2);

  lstm_l1_kernel<<<dim3(16), dim3(128), SMEM1, stream>>>(
      x, Wih1, Whh1, bih1, bhh1, h1buf);
  lstm_l2_head_kernel<<<dim3(16), dim3(128), SMEM2, stream>>>(
      h1buf, Wih2, Whh2, bih2, bhh2, Wl1, bl1, Wl2, bl2, (float*)d_out);
}
